// AttentionCell_21732534518236
// MI455X (gfx1250) — matrix operand order repaired and checked by reading, not compiled, not run on hardware
//
#include <hip/hip_runtime.h>
#include <hip/hip_bf16.h>
#include <math.h>

// ---------------------------------------------------------------------------
// CDNA5 (gfx1250) wave32 WMMA types
// ---------------------------------------------------------------------------
typedef __attribute__((ext_vector_type(16))) __bf16 v16bf;
typedef __attribute__((ext_vector_type(8)))  float  v8f;
typedef __attribute__((ext_vector_type(4)))  float  f32x4;

#define WMMA_BF16(a, b, c) \
  __builtin_amdgcn_wmma_f32_16x16x32_bf16(false, (a), false, (b), (short)0, (c), false, false)

static __device__ __forceinline__ v8f zero8() {
  v8f z;
#pragma unroll
  for (int i = 0; i < 8; ++i) z[i] = 0.0f;
  return z;
}

// Pack 16 f32 (already fetched as 4x float4) into a bf16 fragment register set.
static __device__ __forceinline__ v16bf make_frag(f32x4 q0, f32x4 q1, f32x4 q2, f32x4 q3) {
  v16bf r;
  r[0]  = (__bf16)q0[0]; r[1]  = (__bf16)q0[1]; r[2]  = (__bf16)q0[2]; r[3]  = (__bf16)q0[3];
  r[4]  = (__bf16)q1[0]; r[5]  = (__bf16)q1[1]; r[6]  = (__bf16)q1[2]; r[7]  = (__bf16)q1[3];
  r[8]  = (__bf16)q2[0]; r[9]  = (__bf16)q2[1]; r[10] = (__bf16)q2[2]; r[11] = (__bf16)q2[3];
  r[12] = (__bf16)q3[0]; r[13] = (__bf16)q3[1]; r[14] = (__bf16)q3[2]; r[15] = (__bf16)q3[3];
  return r;
}

// A fragment (16x32, bf16): lane L<16 holds row M0+L, K pairs {0..7,16..23};
// lane L>=16 holds row M0+L-16, K pairs {8..15,24..31}.  (ISA 7.12.2)
static __device__ __forceinline__ v16bf load_frag_a(const float* __restrict__ A, int lda,
                                                    int M0, int K0, int lane) {
  const int r  = lane & 15;
  const int kl = (lane >> 4) << 3;   // 0 or 8
  const float* p = A + (size_t)(M0 + r) * (size_t)lda + (K0 + kl);
  f32x4 q0 = *(const f32x4*)(p + 0);
  f32x4 q1 = *(const f32x4*)(p + 4);
  f32x4 q2 = *(const f32x4*)(p + 16);
  f32x4 q3 = *(const f32x4*)(p + 20);
  return make_frag(q0, q1, q2, q3);
}

// B fragment (32x16, bf16) from row-major W[N][K]: lane L holds column N0+(L&15);
// lanes <16: K={0..7,16..23}, lanes >=16: K={8..15,24..31} (pairs per VGPR).
static __device__ __forceinline__ v16bf load_frag_b(const float* __restrict__ W, int ldb,
                                                    int N0, int K0, int lane) {
  const int n  = N0 + (lane & 15);
  const int kl = (lane >> 4) << 3;   // 0 or 8
  const float* p = W + (size_t)n * (size_t)ldb + (K0 + kl);
  f32x4 q0 = *(const f32x4*)(p + 0);
  f32x4 q1 = *(const f32x4*)(p + 4);
  f32x4 q2 = *(const f32x4*)(p + 16);
  f32x4 q3 = *(const f32x4*)(p + 20);
  return make_frag(q0, q1, q2, q3);
}

static __device__ __forceinline__ float sigmoidf_(float x) {
  return 1.0f / (1.0f + __expf(-x));
}

// ---------------------------------------------------------------------------
// Kernel C: proj_prev = prev_h @ W_h2h^T + b_h2h        [256 x 512]
// block = 8 waves; block tile 32(M) x 512(N); wave tile 32 x 64.
// ---------------------------------------------------------------------------
__global__ void __launch_bounds__(256)
proj_prev_kernel(const float* __restrict__ prev_h, const float* __restrict__ W_h2h,
                 const float* __restrict__ b_h2h, float* __restrict__ out) {
  const int lane = threadIdx.x & 31;
  const int wave = threadIdx.x >> 5;
  const int M0 = blockIdx.x * 32;
  const int N0 = wave * 64;

  v8f acc[2][4];
#pragma unroll
  for (int s = 0; s < 2; ++s)
#pragma unroll
    for (int nt = 0; nt < 4; ++nt) acc[s][nt] = zero8();

  for (int k0 = 0; k0 < 512; k0 += 32) {
    v16bf a0 = load_frag_a(prev_h, 512, M0,      k0, lane);
    v16bf a1 = load_frag_a(prev_h, 512, M0 + 16, k0, lane);
#pragma unroll
    for (int nt = 0; nt < 4; ++nt) {
      v16bf bf = load_frag_b(W_h2h, 512, N0 + nt * 16, k0, lane);
      acc[0][nt] = WMMA_BF16(a0, bf, acc[0][nt]);
      acc[1][nt] = WMMA_BF16(a1, bf, acc[1][nt]);
    }
  }

  const int half = lane >> 4;
#pragma unroll
  for (int nt = 0; nt < 4; ++nt) {
    const int n = N0 + nt * 16 + (lane & 15);
    const float bias = b_h2h[n];
#pragma unroll
    for (int s = 0; s < 2; ++s)
#pragma unroll
      for (int j = 0; j < 8; ++j) {
        const int row = M0 + s * 16 + half * 8 + j;
        out[(size_t)row * 512 + n] = acc[s][nt][j] + bias;
      }
  }
}

// ---------------------------------------------------------------------------
// Kernel A: fused attention scores
//   e[m] = sum_n tanh( (batch_H @ W_i2h^T)[m,n] + proj_prev[b,n] ) * w_score[n]
// block tile 32(M) x 512(N=H), K=IN=512.  proj_H never hits memory.
// ---------------------------------------------------------------------------
__global__ void __launch_bounds__(256)
attn_score_kernel(const float* __restrict__ batch_H, const float* __restrict__ W_i2h,
                  const float* __restrict__ proj_prev, const float* __restrict__ W_score,
                  float* __restrict__ e_out) {
  __shared__ float epart[32];
  const int lane = threadIdx.x & 31;
  const int wave = threadIdx.x >> 5;
  const int M0 = blockIdx.x * 32;      // rows of flattened [B*T]
  const int b  = M0 >> 8;              // T=256: 32-row tile never crosses batch rows
  const int N0 = wave * 64;

  if (threadIdx.x < 32) epart[threadIdx.x] = 0.0f;
  __syncthreads();

  v8f acc[2][4];
#pragma unroll
  for (int s = 0; s < 2; ++s)
#pragma unroll
    for (int nt = 0; nt < 4; ++nt) acc[s][nt] = zero8();

  for (int k0 = 0; k0 < 512; k0 += 32) {
    v16bf a0 = load_frag_a(batch_H, 512, M0,      k0, lane);
    v16bf a1 = load_frag_a(batch_H, 512, M0 + 16, k0, lane);
#pragma unroll
    for (int nt = 0; nt < 4; ++nt) {
      v16bf bf = load_frag_b(W_i2h, 512, N0 + nt * 16, k0, lane);
      acc[0][nt] = WMMA_BF16(a0, bf, acc[0][nt]);
      acc[1][nt] = WMMA_BF16(a1, bf, acc[1][nt]);
    }
  }

  // epilogue: tanh + score dot, reduced over this wave's 64 columns
  const int half = lane >> 4;
  float partial[2][8];
#pragma unroll
  for (int s = 0; s < 2; ++s)
#pragma unroll
    for (int j = 0; j < 8; ++j) partial[s][j] = 0.0f;

#pragma unroll
  for (int nt = 0; nt < 4; ++nt) {
    const int n = N0 + nt * 16 + (lane & 15);
    const float pp  = proj_prev[(size_t)b * 512 + n];
    const float wsc = W_score[n];
#pragma unroll
    for (int j = 0; j < 8; ++j) {
      partial[0][j] += tanhf(acc[0][nt][j] + pp) * wsc;
      partial[1][j] += tanhf(acc[1][nt][j] + pp) * wsc;
    }
  }
  // reduce across the 16 lanes of each half-wave (same row set per half)
#pragma unroll
  for (int off = 1; off < 16; off <<= 1) {
#pragma unroll
    for (int s = 0; s < 2; ++s)
#pragma unroll
      for (int j = 0; j < 8; ++j)
        partial[s][j] += __shfl_xor(partial[s][j], off, 32);
  }
  if ((lane & 15) == 0) {
#pragma unroll
    for (int s = 0; s < 2; ++s)
#pragma unroll
      for (int j = 0; j < 8; ++j)
        atomicAdd(&epart[s * 16 + half * 8 + j], partial[s][j]);
  }
  __syncthreads();
  if (threadIdx.x < 32)
    e_out[(size_t)M0 + threadIdx.x] = epart[threadIdx.x];
}

// ---------------------------------------------------------------------------
// Kernel B: softmax over T + context = alpha . batch_H   (memory-bound)
// one block per batch row b; 256 threads.
// ---------------------------------------------------------------------------
__global__ void __launch_bounds__(256)
softmax_ctx_kernel(const float* __restrict__ e, const float* __restrict__ batch_H,
                   float* __restrict__ alpha_out, float* __restrict__ context) {
  __shared__ float red[256];
  __shared__ float salpha[256];
  const int b = blockIdx.x;
  const int t = threadIdx.x;

  const float v = e[(size_t)b * 256 + t];
  red[t] = v;
  __syncthreads();
  for (int s = 128; s > 0; s >>= 1) {
    if (t < s) red[t] = fmaxf(red[t], red[t + s]);
    __syncthreads();
  }
  const float mx = red[0];
  __syncthreads();
  const float p = __expf(v - mx);
  red[t] = p;
  __syncthreads();
  for (int s = 128; s > 0; s >>= 1) {
    if (t < s) red[t] += red[t + s];
    __syncthreads();
  }
  const float a = p / red[0];
  salpha[t] = a;
  alpha_out[(size_t)b * 256 + t] = a;
  __syncthreads();

  float c0 = 0.0f, c1 = 0.0f;
  const float* bh = batch_H + (size_t)b * 256 * 512;
  for (int tt = 0; tt < 256; ++tt) {
    const float al = salpha[tt];
    c0 = fmaf(al, bh[(size_t)tt * 512 + t],       c0);
    c1 = fmaf(al, bh[(size_t)tt * 512 + 256 + t], c1);
  }
  context[(size_t)b * 512 + t]       = c0;
  context[(size_t)b * 512 + 256 + t] = c1;
}

// ---------------------------------------------------------------------------
// Kernel D: LSTM gates GEMM with virtual concatenation
//   A[m,k] = [context | onehot | prev_h]  (K = 512 + 96 + 512 = 1120)
//   B[n,k] = [W_ih | W_hh]                (N = 4H = 2048)
// Region boundaries (512, 608) are multiples of the K-tile (32), so a
// fragment never straddles regions and branches are wave-uniform
// (EXEC stays all-1s into every WMMA, per ISA 7.12 requirement).
// ---------------------------------------------------------------------------
static __device__ __forceinline__ f32x4 ld4A_cat(const float* __restrict__ ctx,
                                                 const float* __restrict__ oneh,
                                                 const float* __restrict__ ph,
                                                 int row, int k) {
  if (k < 512) return *(const f32x4*)(ctx + (size_t)row * 512 + k);
  if (k < 608) return *(const f32x4*)(oneh + (size_t)row * 96 + (k - 512));
  return *(const f32x4*)(ph + (size_t)row * 512 + (k - 608));
}
static __device__ __forceinline__ f32x4 ld4B_cat(const float* __restrict__ Wih,
                                                 const float* __restrict__ Whh,
                                                 int n, int k) {
  if (k < 608) return *(const f32x4*)(Wih + (size_t)n * 608 + k);
  return *(const f32x4*)(Whh + (size_t)n * 512 + (k - 608));
}
static __device__ __forceinline__ v16bf load_frag_a_cat(const float* ctx, const float* oneh,
                                                        const float* ph, int M0, int K0, int lane) {
  const int row = M0 + (lane & 15);
  const int kb  = K0 + ((lane >> 4) << 3);
  return make_frag(ld4A_cat(ctx, oneh, ph, row, kb),
                   ld4A_cat(ctx, oneh, ph, row, kb + 4),
                   ld4A_cat(ctx, oneh, ph, row, kb + 16),
                   ld4A_cat(ctx, oneh, ph, row, kb + 20));
}
static __device__ __forceinline__ v16bf load_frag_b_cat(const float* Wih, const float* Whh,
                                                        int N0, int K0, int lane) {
  const int n  = N0 + (lane & 15);
  const int kb = K0 + ((lane >> 4) << 3);
  return make_frag(ld4B_cat(Wih, Whh, n, kb),
                   ld4B_cat(Wih, Whh, n, kb + 4),
                   ld4B_cat(Wih, Whh, n, kb + 16),
                   ld4B_cat(Wih, Whh, n, kb + 20));
}

__global__ void __launch_bounds__(256)
gates_kernel(const float* __restrict__ ctx, const float* __restrict__ oneh,
             const float* __restrict__ prev_h,
             const float* __restrict__ W_ih, const float* __restrict__ W_hh,
             const float* __restrict__ b_ih, const float* __restrict__ b_hh,
             float* __restrict__ gates) {
  const int lane = threadIdx.x & 31;
  const int wave = threadIdx.x >> 5;
  const int M0 = blockIdx.x * 32;
  const int N0 = blockIdx.y * 512 + wave * 64;

  v8f acc[2][4];
#pragma unroll
  for (int s = 0; s < 2; ++s)
#pragma unroll
    for (int nt = 0; nt < 4; ++nt) acc[s][nt] = zero8();

  for (int k0 = 0; k0 < 1120; k0 += 32) {
    v16bf a0 = load_frag_a_cat(ctx, oneh, prev_h, M0,      k0, lane);
    v16bf a1 = load_frag_a_cat(ctx, oneh, prev_h, M0 + 16, k0, lane);
#pragma unroll
    for (int nt = 0; nt < 4; ++nt) {
      v16bf bf = load_frag_b_cat(W_ih, W_hh, N0 + nt * 16, k0, lane);
      acc[0][nt] = WMMA_BF16(a0, bf, acc[0][nt]);
      acc[1][nt] = WMMA_BF16(a1, bf, acc[1][nt]);
    }
  }

  const int half = lane >> 4;
#pragma unroll
  for (int nt = 0; nt < 4; ++nt) {
    const int n = N0 + nt * 16 + (lane & 15);
    const float bias = b_ih[n] + b_hh[n];
#pragma unroll
    for (int s = 0; s < 2; ++s)
#pragma unroll
      for (int j = 0; j < 8; ++j) {
        const int row = M0 + s * 16 + half * 8 + j;
        gates[(size_t)row * 2048 + n] = acc[s][nt][j] + bias;
      }
  }
}

// ---------------------------------------------------------------------------
// Kernel E: LSTM elementwise (gate order i, f, g, o)
// ---------------------------------------------------------------------------
__global__ void __launch_bounds__(256)
lstm_elem_kernel(const float* __restrict__ gates, const float* __restrict__ prev_c,
                 float* __restrict__ out) {
  const int idx = blockIdx.x * 256 + threadIdx.x;   // 0 .. 256*512-1
  const int bb = idx >> 9;
  const int h  = idx & 511;
  const float* g = gates + (size_t)bb * 2048;
  const float ig = sigmoidf_(g[h]);
  const float fg = sigmoidf_(g[512 + h]);
  const float gg = tanhf(g[1024 + h]);
  const float og = sigmoidf_(g[1536 + h]);
  const float c  = fg * prev_c[idx] + ig * gg;
  out[idx]          = og * tanhf(c);   // new_h
  out[131072 + idx] = c;               // new_c
}

// ---------------------------------------------------------------------------
// Launch: proj_prev -> attention scores -> softmax/context -> gates -> LSTM
// ---------------------------------------------------------------------------
extern "C" void kernel_launch(void* const* d_in, const int* in_sizes, int n_in,
                              void* d_out, int out_size, void* d_ws, size_t ws_size,
                              hipStream_t stream) {
  (void)in_sizes; (void)n_in; (void)out_size; (void)ws_size;

  const float* prev_h  = (const float*)d_in[0];
  const float* prev_c  = (const float*)d_in[1];
  const float* batch_H = (const float*)d_in[2];
  const float* onehots = (const float*)d_in[3];
  const float* W_i2h   = (const float*)d_in[4];
  const float* W_h2h   = (const float*)d_in[5];
  const float* b_h2h   = (const float*)d_in[6];
  const float* W_score = (const float*)d_in[7];
  const float* W_ih    = (const float*)d_in[8];
  const float* W_hh    = (const float*)d_in[9];
  const float* b_ih    = (const float*)d_in[10];
  const float* b_hh    = (const float*)d_in[11];

  float* out = (float*)d_out;            // [new_h(131072) | new_c(131072) | alpha(65536)]
  float* ws  = (float*)d_ws;
  float* proj_prev = ws;                 // 131072 f32
  float* e         = ws + 131072;        // 65536 f32
  float* context   = ws + 196608;        // 131072 f32
  float* gates     = ws + 327680;        // 524288 f32  (total ~3.4 MB)

  proj_prev_kernel<<<8, 256, 0, stream>>>(prev_h, W_h2h, b_h2h, proj_prev);
  attn_score_kernel<<<2048, 256, 0, stream>>>(batch_H, W_i2h, proj_prev, W_score, e);
  softmax_ctx_kernel<<<256, 256, 0, stream>>>(e, batch_H, out + 262144, context);
  gates_kernel<<<dim3(8, 4), 256, 0, stream>>>(context, onehots, prev_h,
                                               W_ih, W_hh, b_ih, b_hh, gates);
  lstm_elem_kernel<<<512, 256, 0, stream>>>(gates, prev_c, out);
}
